// SelfAttention_89421219103653
// MI455X (gfx1250) — compile-verified, matrix operand order checked by reading, not run on hardware
//
#include <hip/hip_runtime.h>
#include <hip/hip_bf16.h>

// ---------------------------------------------------------------------------
// Self-attention (N=4, S=2048, E=1024, H=16, D=64) for gfx1250 (MI455X).
// f16 WMMA (v_wmma_f32_16x16x32_f16) with f32 accumulation.
// - Projections: double-buffered async-to-LDS tiled GEMM.
// - Attention: flash-style streaming softmax in exp2 domain, double-buffered
//   async K/V staging, transposed-packed mask bits, row sums via ones-WMMA.
// ---------------------------------------------------------------------------

typedef __attribute__((ext_vector_type(16))) _Float16 v16h;
typedef __attribute__((ext_vector_type(8)))  _Float16 v8h;
typedef __attribute__((ext_vector_type(4)))  _Float16 v4h;
typedef __attribute__((ext_vector_type(8)))  float    v8f;
typedef __attribute__((ext_vector_type(4)))  int      v4i;

static constexpr int NB = 4;
static constexpr int SL = 2048;
static constexpr int ED = 1024;
static constexpr int HH = 16;
static constexpr int DH = 64;

// --------------------------- async global->LDS -----------------------------
#if defined(__has_builtin)
#if __has_builtin(__builtin_amdgcn_global_load_async_to_lds_b128)
#define HAVE_ASYNC_LDS 1
#endif
#endif
#ifndef HAVE_ASYNC_LDS
#define HAVE_ASYNC_LDS 0
#endif

typedef __attribute__((address_space(1))) v4i gv4i;   // int4 __device__
typedef __attribute__((address_space(3))) v4i lv4i;   // int4 __shared__

__device__ inline void copy16_g2l(const _Float16* src, _Float16* dst_lds) {
#if HAVE_ASYNC_LDS
  __builtin_amdgcn_global_load_async_to_lds_b128(
      (gv4i*)(void*)const_cast<_Float16*>(src),
      (lv4i*)(void*)dst_lds, 0, 0);
#else
  *(v8h*)dst_lds = *(const v8h*)src;
#endif
}

__device__ inline void wait_async_copies() {
#if HAVE_ASYNC_LDS
  asm volatile("s_wait_asynccnt 0x0" ::: "memory");
#endif
}

// ------------------------------ WMMA helpers -------------------------------
__device__ inline v8f wmma_16x16x32(v16h a, v16h b, v8f c) {
  return __builtin_amdgcn_wmma_f32_16x16x32_f16(
      false, a, false, b, (short)0, c, false, false);
}

// 16x32 f16 fragment (A layout; B symmetric, one column per lane).
__device__ inline v16h load_frag(const _Float16* __restrict__ base, int ld) {
  const int lane = threadIdx.x & 31;
  const int r    = lane & 15;
  const int ko   = (lane & 16) ? 8 : 0;
  const _Float16* p = base + (size_t)r * (size_t)ld + ko;
  union { v16h f; v8h h[2]; } u;
  u.h[0] = *(const v8h*)(p);
  u.h[1] = *(const v8h*)(p + 16);
  return u.f;
}

// --------------------------- f32 -> f16 convert ----------------------------
__global__ __launch_bounds__(256) void cvt_f16(const float4* __restrict__ in,
                                               v4h* __restrict__ out, int n4) {
  int i = blockIdx.x * 256 + threadIdx.x;
  if (i < n4) {
    float4 x = in[i];
    v4h o = {(_Float16)x.x, (_Float16)x.y, (_Float16)x.z, (_Float16)x.w};
    out[i] = o;
  }
}

// ------------- pack mask, transposed, to zero-bitmask words ----------------
// Zt[n][k][q/32] bit (q%32) = (mask[n][0][q][k] == 0)
__global__ __launch_bounds__(256) void mask_pack_t(const int* __restrict__ mask,
                                                   unsigned* __restrict__ zt) {
  const size_t gid = (size_t)blockIdx.x * 256 + threadIdx.x;
  const int    k   = (int)(gid % SL);
  const size_t t1  = gid / SL;
  const int    qw  = (int)(t1 % (SL / 32));
  const int    n   = (int)(t1 / (SL / 32));
  const int* mp = mask + ((size_t)n * SL + (size_t)qw * 32) * SL + k;
  unsigned wd = 0;
#pragma unroll
  for (int i = 0; i < 32; ++i)
    wd |= (unsigned)(mp[(size_t)i * SL] == 0) << i;
  zt[((size_t)n * SL + k) * (SL / 32) + qw] = wd;
}

// ---------------------------------------------------------------------------
// GEMM: C[M x NC] = A[M x K] * B[NC x K]^T with double-buffered async-to-LDS
// staging.  Block = 8 waves (2m x 4n) -> 64 x 256 C tile.
// MODE 0: f16 out. MODE 1: f16 out scattered to V^T [n][h][d][s].
// MODE 2: f32 out + bias.
// ---------------------------------------------------------------------------
template <int MODE>
__global__ __launch_bounds__(256) void gemm16(const _Float16* __restrict__ A,
                                              const _Float16* __restrict__ B,
                                              void* __restrict__ Cout,
                                              const float* __restrict__ bias,
                                              int M, int NC, int K) {
  __shared__ __align__(16) _Float16 Ash[2][64 * 32];
  __shared__ __align__(16) _Float16 Bsh[2][256 * 32];

  const int tid  = threadIdx.x;
  const int w    = tid >> 5;
  const int lane = tid & 31;
  const int wm   = w & 1;
  const int wn   = w >> 1;
  const int mb   = blockIdx.x * 64;
  const int nb   = blockIdx.y * 256;

  const int srow = tid >> 2;            // 0..63
  const int scol = (tid & 3) * 8;       // 0,8,16,24
  const _Float16* Ag = A + (size_t)(mb + srow) * K + scol;
  const _Float16* Bg = B + (size_t)(nb + srow) * K + scol;

  auto stage = [&](int buf, int k0) {
    copy16_g2l(Ag + k0, &Ash[buf][srow * 32 + scol]);
#pragma unroll
    for (int i = 0; i < 4; ++i)
      copy16_g2l(Bg + (size_t)(i * 64) * K + k0,
                 &Bsh[buf][(i * 64 + srow) * 32 + scol]);
  };

  v8f c[2][4] = {{{}, {}, {}, {}}, {{}, {}, {}, {}}};

  stage(0, 0);
  int buf = 0;
  for (int k0 = 0; k0 < K; k0 += 32) {
    wait_async_copies();
    __syncthreads();
    if (k0 + 32 < K) stage(buf ^ 1, k0 + 32);

    v16h a0 = load_frag(&Ash[buf][(wm * 32 +  0) * 32], 32);
    v16h a1 = load_frag(&Ash[buf][(wm * 32 + 16) * 32], 32);
    v16h b0 = load_frag(&Bsh[buf][(wn * 64 +  0) * 32], 32);
    v16h b1 = load_frag(&Bsh[buf][(wn * 64 + 16) * 32], 32);
    v16h b2 = load_frag(&Bsh[buf][(wn * 64 + 32) * 32], 32);
    v16h b3 = load_frag(&Bsh[buf][(wn * 64 + 48) * 32], 32);
    c[0][0] = wmma_16x16x32(a0, b0, c[0][0]);
    c[0][1] = wmma_16x16x32(a0, b1, c[0][1]);
    c[0][2] = wmma_16x16x32(a0, b2, c[0][2]);
    c[0][3] = wmma_16x16x32(a0, b3, c[0][3]);
    c[1][0] = wmma_16x16x32(a1, b0, c[1][0]);
    c[1][1] = wmma_16x16x32(a1, b1, c[1][1]);
    c[1][2] = wmma_16x16x32(a1, b2, c[1][2]);
    c[1][3] = wmma_16x16x32(a1, b3, c[1][3]);

    __syncthreads();
    buf ^= 1;
  }

  const int m0  = mb + wm * 32;
  const int n0  = nb + wn * 64;
  const int cn  = lane & 15;
  const int mro = (lane & 16) ? 8 : 0;
#pragma unroll
  for (int i = 0; i < 2; ++i) {
#pragma unroll
    for (int j = 0; j < 4; ++j) {
#pragma unroll
      for (int r = 0; r < 8; ++r) {
        const int row = m0 + i * 16 + r + mro;
        const int col = n0 + j * 16 + cn;
        if constexpr (MODE == 0) {
          ((_Float16*)Cout)[(size_t)row * NC + col] = (_Float16)c[i][j][r];
        } else if constexpr (MODE == 1) {
          const int nn = row >> 11;
          const int s  = row & (SL - 1);
          const int h  = col >> 6;
          const int d  = col & (DH - 1);
          ((_Float16*)Cout)[(((size_t)nn * HH + h) * DH + d) * SL + s] =
              (_Float16)c[i][j][r];
        } else {
          ((float*)Cout)[(size_t)row * NC + col] = c[i][j][r] + bias[col];
        }
      }
    }
  }
}

// ---------------------------------------------------------------------------
// Flash attention.  8 waves per block share one (n,h); each wave owns one
// 16-query tile.  Softmax runs in the exp2 domain (scale folded into Q).
// Row sums ride along as a 5th WMMA against an all-ones B fragment.
// ---------------------------------------------------------------------------
__global__ __launch_bounds__(256) void attn_flash(
    const _Float16* __restrict__ Q,    // [n][s][h*D+d]
    const _Float16* __restrict__ Kp,   // [n][s][h*D+d]
    const _Float16* __restrict__ Vt,   // [n][h][d][s]
    const unsigned* __restrict__ Zt,   // [n][k][S/32] transposed zero-bitmask
    _Float16* __restrict__ O) {        // [n][s][h*D+d]
  __shared__ __align__(16) _Float16 Ksh[2][32 * 64];    // keys x d
  __shared__ __align__(16) _Float16 Vsh[2][64 * 32];    // d x keys
  __shared__ __align__(16) _Float16 Psh[8 * 16 * 32];   // per-wave P staging

  const int tid  = threadIdx.x;
  const int w    = tid >> 5;
  const int lane = tid & 31;

  constexpr int QBLK = (SL / 16) / 8;              // 16 blocks per (n,h)
  const int qg = blockIdx.x % QBLK;
  const int nh = blockIdx.x / QBLK;
  const int h  = nh % HH;
  const int n  = nh / HH;
  const int q0 = (qg * 8 + w) * 16;

  const int cn  = lane & 15;
  const int mro = (lane & 16) ? 8 : 0;

  // staging assignments (per thread, 16B each for K and V)
  const int krow = tid >> 3;           // 0..31  (key within block)
  const int kcol = (tid & 7) * 8;      // 0..56  (d)
  const int vrow = tid >> 2;           // 0..63  (d)
  const int vcol = (tid & 3) * 8;      // 0..24  (key within block)
  const _Float16* Kg = Kp + ((size_t)n * SL + krow) * ED + h * DH + kcol;
  const _Float16* Vg = Vt + ((size_t)((size_t)n * HH + h) * DH + vrow) * SL + vcol;

  auto stageKV = [&](int buf, int j0) {
    copy16_g2l(Kg + (size_t)j0 * ED, &Ksh[buf][krow * 64 + kcol]);
    copy16_g2l(Vg + j0,              &Vsh[buf][vrow * 32 + vcol]);
  };

  // Q fragments with the 1/sqrt(D) * log2(e) scale folded in (exp2 domain)
  const _Float16* Qb = Q + ((size_t)n * SL + q0) * ED + h * DH;
  v16h qa0 = load_frag(Qb, ED);
  v16h qa1 = load_frag(Qb + 32, ED);
  const _Float16 qs = (_Float16)0.18033688f;   // log2(e)/8
#pragma unroll
  for (int i = 0; i < 16; ++i) { qa0[i] *= qs; qa1[i] *= qs; }

  // all-ones B fragment: row-sum accumulator tile (P * ones)
  v16h bones;
#pragma unroll
  for (int i = 0; i < 16; ++i) bones[i] = (_Float16)1.0f;

  // transposed packed mask: lane reads word for key column, bit = query row
  const unsigned* Ztn = Zt + (size_t)n * SL * (SL / 32);
  const int qw  = q0 >> 5;
  const int zsh = (q0 & 16) + mro;     // bit base for this lane's rows

  v8f acc[5] = {{}, {}, {}, {}, {}};   // acc[4] = running row sums
  float mrow[8];
#pragma unroll
  for (int r = 0; r < 8; ++r) mrow[r] = -1e30f;

  _Float16* pw = Psh + w * (16 * 32);

  stageKV(0, 0);
  int buf = 0;
  for (int j0 = 0; j0 < SL; j0 += 32) {
    wait_async_copies();
    __syncthreads();                           // K/V for this step staged
    if (j0 + 32 < SL) stageKV(buf ^ 1, j0 + 32);

    // --- scores (exp2 domain): S(16x32) = Qs(16x64) * K_blk(32x64)^T -----
    const _Float16* Kb = &Ksh[buf][0];
    v8f s0 = {}, s1 = {};
    s0 = wmma_16x16x32(qa0, load_frag(Kb, 64), s0);
    s0 = wmma_16x16x32(qa1, load_frag(Kb + 32, 64), s0);
    s1 = wmma_16x16x32(qa0, load_frag(Kb + 16 * 64, 64), s1);
    s1 = wmma_16x16x32(qa1, load_frag(Kb + 16 * 64 + 32, 64), s1);

    // --- mask via transposed bitmask: 2 b32 loads, no shuffles ------------
    const unsigned z0 = Ztn[(size_t)(j0 + cn) * (SL / 32) + qw];
    const unsigned z1 = Ztn[(size_t)(j0 + cn + 16) * (SL / 32) + qw];
#pragma unroll
    for (int r = 0; r < 8; ++r) {
      if ((z0 >> (zsh + r)) & 1) s0[r] = -1e30f;
      if ((z1 >> (zsh + r)) & 1) s1[r] = -1e30f;
    }

    // --- online softmax (base-2) ------------------------------------------
    float nmax[8], alpha[8];
#pragma unroll
    for (int r = 0; r < 8; ++r) {
      float v = fmaxf(s0[r], s1[r]);
#pragma unroll
      for (int off = 1; off < 16; off <<= 1)
        v = fmaxf(v, __shfl_xor(v, off, 32));
      nmax[r]  = fmaxf(mrow[r], v);
      alpha[r] = __builtin_amdgcn_exp2f(mrow[r] - nmax[r]);
      mrow[r]  = nmax[r];
    }
#pragma unroll
    for (int r = 0; r < 8; ++r) {
      s0[r] = __builtin_amdgcn_exp2f(s0[r] - nmax[r]);
      s1[r] = __builtin_amdgcn_exp2f(s1[r] - nmax[r]);
    }
#pragma unroll
    for (int dt = 0; dt < 5; ++dt)
#pragma unroll
      for (int r = 0; r < 8; ++r) acc[dt][r] *= alpha[r];

    // --- restage P (C layout -> A fragment) through LDS -------------------
#pragma unroll
    for (int r = 0; r < 8; ++r) {
      pw[(r + mro) * 32 + cn]      = (_Float16)s0[r];
      pw[(r + mro) * 32 + 16 + cn] = (_Float16)s1[r];
    }
    const v16h pa = load_frag(pw, 32);   // same-wave DS ops are in-order

    // --- O += P*V; row sums += P*1 ----------------------------------------
#pragma unroll
    for (int dt = 0; dt < 4; ++dt)
      acc[dt] = wmma_16x16x32(pa, load_frag(&Vsh[buf][dt * 16 * 32], 32), acc[dt]);
    acc[4] = wmma_16x16x32(pa, bones, acc[4]);

    __syncthreads();                          // all reads of buf done
    buf ^= 1;
  }

  // --- normalize and write out ---------------------------------------------
  float inv[8];
#pragma unroll
  for (int r = 0; r < 8; ++r) inv[r] = __builtin_amdgcn_rcpf(acc[4][r]);
#pragma unroll
  for (int dt = 0; dt < 4; ++dt) {
#pragma unroll
    for (int r = 0; r < 8; ++r) {
      const float o = acc[dt][r] * inv[r];
      const size_t oi =
          ((size_t)n * SL + q0 + r + mro) * ED + h * DH + dt * 16 + cn;
      O[oi] = (_Float16)o;
    }
  }
}

// ---------------------------------------------------------------------------
// Host-side launcher
// ---------------------------------------------------------------------------
extern "C" void kernel_launch(void* const* d_in, const int* in_sizes, int n_in,
                              void* d_out, int out_size, void* d_ws, size_t ws_size,
                              hipStream_t stream) {
  (void)in_sizes; (void)n_in; (void)out_size; (void)ws_size;

  const float* values = (const float*)d_in[0];
  const float* keys   = (const float*)d_in[1];
  const float* query  = (const float*)d_in[2];
  const int*   mask   = (const int*)d_in[3];
  const float* Wv     = (const float*)d_in[4];
  const float* Wk     = (const float*)d_in[5];
  const float* Wq     = (const float*)d_in[6];
  const float* Wo     = (const float*)d_in[7];
  const float* bo     = (const float*)d_in[8];
  float*       out    = (float*)d_out;

  const size_t NSE = (size_t)NB * SL * ED;  // 8 Mi elems
  const size_t EE  = (size_t)ED * ED;       // 1 Mi elems
  const size_t NSS = (size_t)NB * SL * SL;  // mask elements

  char* ws = (char*)d_ws;
  _Float16* Xv16 = (_Float16*)ws;            ws += NSE * 2;
  _Float16* Xk16 = (_Float16*)ws;            ws += NSE * 2;
  _Float16* Xq16 = (_Float16*)ws;            ws += NSE * 2;
  _Float16* Wv16 = (_Float16*)ws;            ws += EE * 2;
  _Float16* Wk16 = (_Float16*)ws;            ws += EE * 2;
  _Float16* Wq16 = (_Float16*)ws;            ws += EE * 2;
  _Float16* Wo16 = (_Float16*)ws;            ws += EE * 2;
  _Float16* Q16  = (_Float16*)ws;            ws += NSE * 2;
  _Float16* K16  = (_Float16*)ws;            ws += NSE * 2;
  _Float16* Vt16 = (_Float16*)ws;            ws += NSE * 2;
  _Float16* AO16 = (_Float16*)ws;            ws += NSE * 2;
  unsigned* Zt   = (unsigned*)ws;            ws += (NSS / 32) * 4;

  // 1) f32 -> f16 conversions (vectorized x4) + transposed mask packing
  const int cb = 256;
  const int nse4 = (int)(NSE / 4), ee4 = (int)(EE / 4);
  cvt_f16<<<(nse4 + cb - 1) / cb, cb, 0, stream>>>((const float4*)values, (v4h*)Xv16, nse4);
  cvt_f16<<<(nse4 + cb - 1) / cb, cb, 0, stream>>>((const float4*)keys,   (v4h*)Xk16, nse4);
  cvt_f16<<<(nse4 + cb - 1) / cb, cb, 0, stream>>>((const float4*)query,  (v4h*)Xq16, nse4);
  cvt_f16<<<(ee4  + cb - 1) / cb, cb, 0, stream>>>((const float4*)Wv, (v4h*)Wv16, ee4);
  cvt_f16<<<(ee4  + cb - 1) / cb, cb, 0, stream>>>((const float4*)Wk, (v4h*)Wk16, ee4);
  cvt_f16<<<(ee4  + cb - 1) / cb, cb, 0, stream>>>((const float4*)Wq, (v4h*)Wq16, ee4);
  cvt_f16<<<(ee4  + cb - 1) / cb, cb, 0, stream>>>((const float4*)Wo, (v4h*)Wo16, ee4);
  mask_pack_t<<<(int)(NSS / 32 / cb), cb, 0, stream>>>(mask, Zt);

  // 2) projections: [8192 x 1024] = [8192 x 1024] @ [1024 x 1024]^T
  const int M = NB * SL;
  dim3 ggrid(M / 64, ED / 256);
  gemm16<0><<<ggrid, 256, 0, stream>>>(Xq16, Wq16, (void*)Q16,  nullptr, M, ED, ED);
  gemm16<0><<<ggrid, 256, 0, stream>>>(Xk16, Wk16, (void*)K16,  nullptr, M, ED, ED);
  gemm16<1><<<ggrid, 256, 0, stream>>>(Xv16, Wv16, (void*)Vt16, nullptr, M, ED, ED);

  // 3) flash attention: 1024 blocks x 8 waves
  const int nblk = NB * HH * ((SL / 16) / 8);
  attn_flash<<<nblk, 256, 0, stream>>>(Q16, K16, Vt16, Zt, AO16);

  // 4) output projection + bias -> f32
  gemm16<2><<<ggrid, 256, 0, stream>>>(AO16, Wo16, (void*)out, bo, M, ED, ED);
}
